// PatchedDbrxAttention_29240137351598
// MI455X (gfx1250) — matrix operand order repaired and checked by reading, not compiled, not run on hardware
//
#include <hip/hip_runtime.h>

#define T_TOK 8192
#define S_LEN 1024
#define DMODEL 4096
#define NH 32
#define NKV 8
#define HD 128
#define QKVD 6144            // (32 + 2*8) * 128
#define KOFF 4096            // 32*128
#define VOFF 5120            // 40*128
#define CLIPV 8.0f

typedef _Float16 h8  __attribute__((ext_vector_type(8)));
typedef _Float16 v16h __attribute__((ext_vector_type(16)));
typedef float    v8f  __attribute__((ext_vector_type(8)));

union Frag {
    v16h v;
    h8   h[2];
    _Float16 e[16];
};

// ---- CDNA5 async memory->LDS copy (ASYNCcnt-tracked), §15.18.3 op 98 ------
__device__ __forceinline__ void async_load16(unsigned int lds_off, const void* gaddr) {
    asm volatile("global_load_async_to_lds_b128 %0, %1, off"
                 :: "v"(lds_off), "v"((unsigned long long)(uintptr_t)gaddr)
                 : "memory");
}
__device__ __forceinline__ void wait_async0() {
    asm volatile("s_wait_asynccnt 0x0" ::: "memory");
}

// ---------------------------------------------------------------- converts
__global__ __launch_bounds__(256) void cvt_kernel(const float* __restrict__ src,
                                                  _Float16* __restrict__ dst,
                                                  size_t n) {
    size_t i = (size_t)blockIdx.x * blockDim.x + threadIdx.x;
    if (i < n) dst[i] = (_Float16)src[i];
}

// ---------------------------------------------------------------- RoPE (q + k heads, contiguous heads 0..39)
__global__ __launch_bounds__(256) void rope_kernel(_Float16* __restrict__ qkv,
                                                   const int* __restrict__ pos) {
    const size_t idx = (size_t)blockIdx.x * blockDim.x + threadIdx.x;
    const size_t total = (size_t)T_TOK * 40 * 64;
    if (idx >= total) return;
    const int d  = (int)(idx & 63);
    const int hh = (int)((idx >> 6) % 40);
    const int t  = (int)(idx / (40 * 64));
    const float p = (float)pos[t];
    const float inv_freq = __powf(10000.0f, -(float)d * (1.0f / 64.0f));
    const float fr = p * inv_freq;
    float sn, cs;
    __sincosf(fr, &sn, &cs);
    _Float16* base = qkv + (size_t)t * QKVD + (size_t)hh * HD;
    const float x1 = (float)base[d];
    const float x2 = (float)base[d + 64];
    base[d]      = (_Float16)(x1 * cs - x2 * sn);
    base[d + 64] = (_Float16)(x2 * cs + x1 * sn);
}

// ---------------------------------------------------------------- fragment loaders
__device__ __forceinline__ void load_afrag(Frag& f, const _Float16* rowp, int k0, int l16) {
    const _Float16* p = rowp + k0 + l16 * 16;
    f.h[0] = *(const h8*)(p);
    f.h[1] = *(const h8*)(p + 8);
}
__device__ __forceinline__ void load_bfrag(Frag& f, const _Float16* rowp, int k0, int l16) {
    const _Float16* p = rowp + k0 + l16 * 16;
    f.h[0] = *(const h8*)(p);
    f.h[1] = *(const h8*)(p + 8);
}

// ---------------------------------------------------------------- WMMA GEMM: C[M,N] = A[M,K] @ W[N,K]^T
// One wave computes a 32x64 tile (8 accumulators). K loop is unrolled by 2
// with two statically-named fragment buffers so double buffering stays in
// VGPRs (no runtime-indexed arrays -> no scratch/LDS demotion): loads for the
// next 32-wide half-step issue before the 8 WMMAs of the current one.
__global__ __launch_bounds__(256) void gemm_wmma_kernel(const _Float16* __restrict__ A,
                                                        const _Float16* __restrict__ W,
                                                        int M, int N, int K,
                                                        float clipv,
                                                        _Float16* __restrict__ outH,
                                                        float* __restrict__ outF) {
    const int wid  = (blockIdx.x * blockDim.x + threadIdx.x) >> 5;
    const int lane = threadIdx.x & 31;
    const int ln   = lane & 15;
    const int l16  = lane >> 4;
    const int ntiles = N >> 6;
    const int mt = wid / ntiles;
    const int nt = wid % ntiles;
    if (mt >= (M >> 5)) return;
    const int m0 = mt << 5;
    const int n0 = nt << 6;

    const _Float16* arow0 = A + (size_t)(m0 + ln) * K;
    const _Float16* arow1 = arow0 + (size_t)16 * K;
    const _Float16* wrow[4];
#pragma unroll
    for (int j = 0; j < 4; ++j) wrow[j] = W + (size_t)(n0 + j * 16 + ln) * K;

    const v8f zero = {};
    v8f acc[2][4];
#pragma unroll
    for (int mi = 0; mi < 2; ++mi)
#pragma unroll
        for (int j = 0; j < 4; ++j) acc[mi][j] = zero;

    Frag a0[2], a1[2], b0[4], b1[4];
    load_afrag(a0[0], arow0, 0, l16);
    load_afrag(a0[1], arow1, 0, l16);
#pragma unroll
    for (int j = 0; j < 4; ++j) load_bfrag(b0[j], wrow[j], 0, l16);

    // K is a multiple of 64 for every GEMM in this pipeline.
    for (int k0 = 0; k0 < K; k0 += 64) {
        // stage half-step k0+32 into buffer 1
        load_afrag(a1[0], arow0, k0 + 32, l16);
        load_afrag(a1[1], arow1, k0 + 32, l16);
#pragma unroll
        for (int j = 0; j < 4; ++j) load_bfrag(b1[j], wrow[j], k0 + 32, l16);
        if (k0 + 256 < K) {  // pull streams ahead into cache (global_prefetch_b8)
            __builtin_prefetch(arow0 + k0 + 256, 0, 1);
            __builtin_prefetch(arow1 + k0 + 256, 0, 1);
            __builtin_prefetch(wrow[0] + k0 + 256, 0, 1);
            __builtin_prefetch(wrow[1] + k0 + 256, 0, 1);
            __builtin_prefetch(wrow[2] + k0 + 256, 0, 1);
            __builtin_prefetch(wrow[3] + k0 + 256, 0, 1);
        }
        // consume buffer 0 (k0)
#pragma unroll
        for (int mi = 0; mi < 2; ++mi)
#pragma unroll
            for (int j = 0; j < 4; ++j)
                acc[mi][j] = __builtin_amdgcn_wmma_f32_16x16x32_f16(
                    false, (mi ? a0[1].v : a0[0].v), false, b0[j].v,
                    (short)0, acc[mi][j], false, false);
        // stage full-step k0+64 into buffer 0
        if (k0 + 64 < K) {
            load_afrag(a0[0], arow0, k0 + 64, l16);
            load_afrag(a0[1], arow1, k0 + 64, l16);
#pragma unroll
            for (int j = 0; j < 4; ++j) load_bfrag(b0[j], wrow[j], k0 + 64, l16);
        }
        // consume buffer 1 (k0+32)
#pragma unroll
        for (int mi = 0; mi < 2; ++mi)
#pragma unroll
            for (int j = 0; j < 4; ++j)
                acc[mi][j] = __builtin_amdgcn_wmma_f32_16x16x32_f16(
                    false, (mi ? a1[1].v : a1[0].v), false, b1[j].v,
                    (short)0, acc[mi][j], false, false);
    }

#pragma unroll
    for (int mi = 0; mi < 2; ++mi)
#pragma unroll
        for (int j = 0; j < 4; ++j) {
            const int col = n0 + j * 16 + ln;
#pragma unroll
            for (int i = 0; i < 8; ++i) {
                const size_t r = (size_t)(m0 + mi * 16 + i + l16 * 8);
                float vv = acc[mi][j][i];
                if (outH) {
                    vv = fminf(fmaxf(vv, -clipv), clipv);
                    outH[r * (size_t)N + col] = (_Float16)vv;
                } else {
                    outF[r * (size_t)N + col] = vv;
                }
            }
        }
}

// ---------------------------------------------------------------- flash attention
// One wave: (b, head, 16-query tile). Keys in 32-wide tiles; V tile is staged
// into LDS with async memory->LDS copies that overlap the QK^T WMMAs and the
// softmax, waited on (ASYNCcnt) just before the PV gather.
__global__ __launch_bounds__(128) void attn_kernel(const _Float16* __restrict__ qkv,
                                                   _Float16* __restrict__ attnOut) {
    __shared__ __align__(16) _Float16 lds[4 * (32 * HD + 16 * 32)];
    const int lane = threadIdx.x & 31;
    const int wib  = threadIdx.x >> 5;
    const int gw   = blockIdx.x * 4 + wib;
    const int qt = gw & 63;
    const int h  = (gw >> 6) & 31;
    const int b  = gw >> 11;
    const int ln  = lane & 15;
    const int l16 = lane >> 4;
    const int q0 = qt * 16;
    const int kv = h >> 2;
    const float scale = 0.08838834764831845f;  // 1/sqrt(128)

    _Float16* vbuf = lds + wib * (32 * HD + 512);
    _Float16* pbuf = vbuf + 32 * HD;
    const unsigned int vbuf_lds = (unsigned int)(uintptr_t)(vbuf + lane * HD);

    // Q fragments (16 queries x 128 dims = 4 A-fragments), kept in registers
    Frag qf[4];
    {
        const _Float16* qp = qkv + (size_t)(b * S_LEN + q0 + ln) * QKVD + (size_t)h * HD;
#pragma unroll
        for (int c = 0; c < 4; ++c) {
            const _Float16* p = qp + c * 32 + l16 * 16;
            qf[c].h[0] = *(const h8*)(p);
            qf[c].h[1] = *(const h8*)(p + 8);
        }
    }

    const v8f zero = {};
    v8f o[8];
    float m[8], lsum[8];
#pragma unroll
    for (int d = 0; d < 8; ++d) o[d] = zero;
#pragma unroll
    for (int i = 0; i < 8; ++i) { m[i] = -1e30f; lsum[i] = 0.0f; }

    const int ktiles = (q0 + 15) / 32 + 1;
    for (int kt = 0; kt < ktiles; ++kt) {
        const int kbase = kt * 32;

        // async-stage V tile (32 keys x 128 dims) into LDS: lane r copies key row r
        {
            const _Float16* vp = qkv + (size_t)(b * S_LEN + kbase + lane) * QKVD
                                 + VOFF + (size_t)kv * HD;
#pragma unroll
            for (int c = 0; c < 16; ++c)
                async_load16(vbuf_lds + c * 16, vp + c * 8);
        }

        // S = Q @ K^T  (two 16x16 column halves, 4 head-dim chunks each)
        v8f s0 = zero, s1 = zero;
        {
            const _Float16* kp0 = qkv + (size_t)(b * S_LEN + kbase + ln) * QKVD
                                  + KOFF + (size_t)kv * HD;
            const _Float16* kp1 = kp0 + (size_t)16 * QKVD;
#pragma unroll
            for (int c = 0; c < 4; ++c) {
                Frag bf;
                const _Float16* p = kp0 + c * 32 + l16 * 16;
                bf.h[0] = *(const h8*)(p);
                bf.h[1] = *(const h8*)(p + 8);
                s0 = __builtin_amdgcn_wmma_f32_16x16x32_f16(
                    false, qf[c].v, false, bf.v, (short)0, s0, false, false);
            }
#pragma unroll
            for (int c = 0; c < 4; ++c) {
                Frag bf;
                const _Float16* p = kp1 + c * 32 + l16 * 16;
                bf.h[0] = *(const h8*)(p);
                bf.h[1] = *(const h8*)(p + 8);
                s1 = __builtin_amdgcn_wmma_f32_16x16x32_f16(
                    false, qf[c].v, false, bf.v, (short)0, s1, false, false);
            }
        }

        // online softmax (row i lives in VGPR i across one 16-lane half)
#pragma unroll
        for (int i = 0; i < 8; ++i) {
            const int qi  = q0 + i + l16 * 8;
            const int k0i = kbase + ln;
            float a0 = (k0i <= qi)      ? s0[i] * scale : -1e30f;
            float a1 = (k0i + 16 <= qi) ? s1[i] * scale : -1e30f;
            float r = fmaxf(a0, a1);
            r = fmaxf(r, __shfl_xor(r, 1, 32));
            r = fmaxf(r, __shfl_xor(r, 2, 32));
            r = fmaxf(r, __shfl_xor(r, 4, 32));
            r = fmaxf(r, __shfl_xor(r, 8, 32));
            const float mn    = fmaxf(m[i], r);
            const float alpha = __expf(m[i] - mn);
            const float p0 = __expf(a0 - mn);
            const float p1 = __expf(a1 - mn);
            float rs = p0 + p1;
            rs += __shfl_xor(rs, 1, 32);
            rs += __shfl_xor(rs, 2, 32);
            rs += __shfl_xor(rs, 4, 32);
            rs += __shfl_xor(rs, 8, 32);
            lsum[i] = lsum[i] * alpha + rs;
            m[i] = mn;
#pragma unroll
            for (int d = 0; d < 8; ++d) o[d][i] *= alpha;
            const int mr = i + l16 * 8;
            pbuf[mr * 32 + ln]      = (_Float16)p0;
            pbuf[mr * 32 + 16 + ln] = (_Float16)p1;
        }

        // reload P as an A fragment (16x32, K = 32 keys)
        Frag pf;
        {
            const _Float16* pp = pbuf + ln * 32 + l16 * 16;
            pf.h[0] = *(const h8*)(pp);
            pf.h[1] = *(const h8*)(pp + 8);
        }

        // wait for the async V staging, then O += P @ V
        wait_async0();
#pragma unroll
        for (int d = 0; d < 8; ++d) {
            Frag vf;
            const int n  = d * 16 + ln;
            const int kb = l16 * 16;
#pragma unroll
            for (int kk = 0; kk < 16; ++kk)
                vf.e[kk] = vbuf[(kb + kk) * HD + n];
            o[d] = __builtin_amdgcn_wmma_f32_16x16x32_f16(
                false, pf.v, false, vf.v, (short)0, o[d], false, false);
        }
    }

    // epilogue: O / l -> f16 attn buffer [T, 4096]
#pragma unroll
    for (int i = 0; i < 8; ++i) {
        const float inv = 1.0f / lsum[i];
        const size_t r = (size_t)(b * S_LEN + q0 + i + l16 * 8);
#pragma unroll
        for (int d = 0; d < 8; ++d)
            attnOut[r * DMODEL + (size_t)h * HD + d * 16 + ln] = (_Float16)(o[d][i] * inv);
    }
}

// ---------------------------------------------------------------- launch
extern "C" void kernel_launch(void* const* d_in, const int* in_sizes, int n_in,
                              void* d_out, int out_size, void* d_ws, size_t ws_size,
                              hipStream_t stream) {
    const float* hs   = (const float*)d_in[0];
    const float* wqkv = (const float*)d_in[1];
    const float* wout = (const float*)d_in[2];
    const int*   pos  = (const int*)d_in[3];
    // d_in[4..6] (block_offsets, k_cache, v_cache) are an identity round-trip
    // in the reference and do not affect the output -> unused.
    float* out = (float*)d_out;

    _Float16* hs_h   = (_Float16*)d_ws;                       // T*4096
    _Float16* wqkv_h = hs_h   + (size_t)T_TOK * DMODEL;       // 6144*4096
    _Float16* wout_h = wqkv_h + (size_t)QKVD * DMODEL;        // 4096*4096
    _Float16* qkv_h  = wout_h + (size_t)DMODEL * DMODEL;      // T*6144
    _Float16* attn_h = qkv_h  + (size_t)T_TOK * QKVD;         // T*4096

    {
        size_t n = (size_t)T_TOK * DMODEL;
        cvt_kernel<<<(unsigned)((n + 255) / 256), 256, 0, stream>>>(hs, hs_h, n);
        n = (size_t)QKVD * DMODEL;
        cvt_kernel<<<(unsigned)((n + 255) / 256), 256, 0, stream>>>(wqkv, wqkv_h, n);
        n = (size_t)DMODEL * DMODEL;
        cvt_kernel<<<(unsigned)((n + 255) / 256), 256, 0, stream>>>(wout, wout_h, n);
    }
    {   // qkv = clip(hs @ Wqkv^T), f16 out
        int waves = (T_TOK / 32) * (QKVD / 64);
        gemm_wmma_kernel<<<waves / 8, 256, 0, stream>>>(
            hs_h, wqkv_h, T_TOK, QKVD, DMODEL, CLIPV, qkv_h, nullptr);
    }
    {   // RoPE on q + k heads
        size_t n = (size_t)T_TOK * 40 * 64;
        rope_kernel<<<(unsigned)((n + 255) / 256), 256, 0, stream>>>(qkv_h, pos);
    }
    {   // flash attention, f16 out
        int waves = 8 * 32 * 64;  // B * N_HEADS * (S/16)
        attn_kernel<<<waves / 4, 128, 0, stream>>>(qkv_h, attn_h);
    }
    {   // out = attn @ Wout^T, fp32 out
        int waves = (T_TOK / 32) * (DMODEL / 64);
        gemm_wmma_kernel<<<waves / 8, 256, 0, stream>>>(
            attn_h, wout_h, T_TOK, DMODEL, DMODEL, 0.0f, nullptr, out);
    }
}
